// DSAAttention_43731357008371
// MI455X (gfx1250) — matrix operand order repaired and checked by reading, not compiled, not run on hardware
//
#include <hip/hip_runtime.h>
#include <hip/hip_bf16.h>
#include <cstdint>

// ---------------------------------------------------------------------------
// DSA sparse attention for MI455X (gfx1250, wave32, WMMA bf16 -> f32 acc)
// B=4, T=8192, D=1024, H=16, HD=64, TOP_K=64
// GEMMs use the Tensor Data Mover (tensor_load_to_lds, 6-arg builtin) for
// double-buffered LDS tile staging.
// ---------------------------------------------------------------------------

typedef __attribute__((ext_vector_type(16))) __bf16 v16bf;
typedef __attribute__((ext_vector_type(8)))  __bf16 v8bf;
typedef __attribute__((ext_vector_type(4)))  __bf16 v4bf;
typedef __attribute__((ext_vector_type(8)))  float  v8f;
typedef __attribute__((ext_vector_type(4)))  unsigned int u32x4;
typedef __attribute__((ext_vector_type(8)))  int i32x8;
typedef __attribute__((ext_vector_type(4)))  int i32x4;

#define B_   4
#define T_   8192
#define D_   1024
#define H_   16
#define HD_  64
#define TK_  64
#define M_   (B_ * T_)   // 32768 rows

#if defined(__gfx1250__) && __has_builtin(__builtin_amdgcn_tensor_load_to_lds) && \
    __has_builtin(__builtin_amdgcn_s_wait_tensorcnt)
#define USE_TDM 1
#else
#define USE_TDM 0
#endif

// ---- fragment loaders -----------------------------------------------------
// A-matrix 16x32 bf16 fragment (ISA 7.12.2): lane r=lane&15 holds row r.
// koff = (lane>>4)*8.  elems 0..7  <- K = koff+0..7   (bytes p)
//                       elems 8..15 <- K = koff+16..23 (bytes p+32)
__device__ __forceinline__ v16bf ld_fragA(const __bf16* p) {
  union { v16bf v; v8bf h[2]; } u;
  u.h[0] = *(const v8bf*)(p);
  u.h[1] = *(const v8bf*)(p + 16);
  return u.v;
}
// B-matrix 32x16 bf16 fragment: lane n=lane&15 holds COLUMN n of B,
// koff = (lane>>4)*8, elems 0..7 <- K = koff+0..7, elems 8..15 <- K = koff+16..23.
__device__ __forceinline__ v16bf ld_fragB(const __bf16* p) {
  union { v16bf v; v8bf h[2]; } u;
  u.h[0] = *(const v8bf*)(p);
  u.h[1] = *(const v8bf*)(p + 16);
  return u.v;
}

__device__ __forceinline__ v8f v8f_zero() {
  v8f z = {0.f, 0.f, 0.f, 0.f, 0.f, 0.f, 0.f, 0.f};
  return z;
}

#if USE_TDM
// Issue one TDM 2-D tile load: 128 rows x 64 bf16 columns from a row-major
// tensor with row stride `kdim` elements, into LDS with a 16B pad inserted
// after every 128B row (32 DWORDs) -> LDS row stride 144B (=72 bf16).
// D# layout per CDNA5 ISA.
// 6-arg builtin form: (g0, g1, g2, g3, extra, cpol) with groups 2/3 zeroed
// (2-D tensor -> VADDR2/3 disabled).
__device__ __forceinline__ void tdm_load_tile(uint64_t gaddr, unsigned lds_off,
                                              unsigned kdim, unsigned rows) {
  u32x4 g0;
  g0[0] = 1u;                                     // count=1, user mode
  g0[1] = lds_off;                                // lds_addr (bytes)
  g0[2] = (unsigned)gaddr;                        // global_addr[31:0]
  g0[3] = (unsigned)((gaddr >> 32) & 0x1FFFFFFu)  // global_addr[56:32]
          | 0x80000000u;                          // type=2 ("image")
  i32x8 g1;
  g1[0] = (int)((1u << 16)        // data_size = 2 bytes
              | (1u << 20)        // pad_enable
              | (4u << 22)        // pad_interval: 2^(4+1)=32 DWORDs (128B)
              | (3u << 25));      // pad_amount: 3+1 = 4 DWORDs (16B)
  g1[1] = (int)((kdim & 0xFFFFu) << 16);                  // tensor_dim0[15:0]
  g1[2] = (int)((kdim >> 16) | ((rows & 0xFFFFu) << 16)); // dim0 hi | dim1 lo
  g1[3] = (int)((rows >> 16) | (64u << 16));              // dim1 hi | tile_dim0=64
  g1[4] = (int)128u;                                      // tile_dim1=128, tile_dim2=0
  g1[5] = (int)kdim;                                      // tensor_dim0_stride[31:0]
  g1[6] = 0;                                              // stride hi | dim1_stride lo
  g1[7] = 0;
  i32x4 z4 = {0, 0, 0, 0};
  i32x8 z8 = {0, 0, 0, 0, 0, 0, 0, 0};
  __builtin_amdgcn_tensor_load_to_lds(g0, g1, z4, z4, z8, 0);
}
#endif

// ---- 1) f32 -> bf16 conversion -------------------------------------------
__global__ __launch_bounds__(256) void cvt_bf16_kernel(
    const float* __restrict__ in, __bf16* __restrict__ out, int n4) {
  int i = blockIdx.x * 256 + threadIdx.x;
  if (i < n4) {
    float4 f = ((const float4*)in)[i];
    v4bf o;
    o[0] = (__bf16)f.x; o[1] = (__bf16)f.y;
    o[2] = (__bf16)f.z; o[3] = (__bf16)f.w;
    ((v4bf*)out)[i] = o;
  }
}

// ---- 2) GEMM  C[M,N] = A[M,K] * W[N,K]^T  (both row-major, contiguous K) --
// Block tile 128x128, BK=64.  8 waves in a 2x4 grid; each wave owns 64x32
// (4 M-subtiles x 2 N-subtiles of 16x16), 16 WMMAs per stage.
// TDM double-buffers the LDS tiles: wave0 issues the next stage's DMA before
// computing, waits TENSORcnt after, then the whole block flips buffers.
template <bool OUT_BF16>
__global__ __launch_bounds__(256) void gemm_nt_bf16_kernel(
    const __bf16* __restrict__ A, const __bf16* __restrict__ W,
    void* __restrict__ out, int M, int N, int Kdim) {
  constexpr int BM = 128, BN = 128, BK = 64;
  constexpr int LDT = 72;                 // 144B LDS row stride (128B + 16B pad)
  constexpr int BUFELE = BM * LDT;        // 9216 bf16 = 18432 B per buffer
  __shared__ __align__(16) __bf16 sA[2 * BUFELE];
  __shared__ __align__(16) __bf16 sB[2 * BUFELE];

  const int tid  = threadIdx.x;
  const int wave = tid >> 5;
  const int lane = tid & 31;
  const int bm = blockIdx.x * BM;
  const int bn = blockIdx.y * BN;
  const int wm = (wave >> 2) * 64;   // 0 or 64
  const int wn = (wave & 3) * 32;    // 0,32,64,96
  const int r     = lane & 15;
  const int koff  = (lane >> 4) * 8;   // A-frag K offset
  const int kbase = (lane >> 4) * 16;  // unused

  v8f acc[4][2];
#pragma unroll
  for (int i = 0; i < 4; ++i)
#pragma unroll
    for (int j = 0; j < 2; ++j) acc[i][j] = v8f_zero();

#if USE_TDM
  const unsigned ldsA = (unsigned)(uintptr_t)(void*)&sA[0];
  const unsigned ldsB = (unsigned)(uintptr_t)(void*)&sB[0];
  const uint64_t gbaseA = (uint64_t)(uintptr_t)A + (uint64_t)bm * Kdim * 2;
  const uint64_t gbaseB = (uint64_t)(uintptr_t)W + (uint64_t)bn * Kdim * 2;
  if (wave == 0) {
    tdm_load_tile(gbaseA, ldsA, (unsigned)Kdim, (unsigned)BM);
    tdm_load_tile(gbaseB, ldsB, (unsigned)Kdim, (unsigned)BN);
    __builtin_amdgcn_s_wait_tensorcnt(0);
  }
  __syncthreads();
#endif

  int buf = 0;
  for (int kt = 0; kt < Kdim; kt += BK) {
#if USE_TDM
    if (wave == 0 && (kt + BK) < Kdim) {   // prefetch next stage into buf^1
      tdm_load_tile(gbaseA + (uint64_t)(kt + BK) * 2,
                    ldsA + (unsigned)((buf ^ 1) * BUFELE * 2),
                    (unsigned)Kdim, (unsigned)BM);
      tdm_load_tile(gbaseB + (uint64_t)(kt + BK) * 2,
                    ldsB + (unsigned)((buf ^ 1) * BUFELE * 2),
                    (unsigned)Kdim, (unsigned)BN);
    }
#else
    // cooperative fallback load: 128 rows x 64 bf16 = 1024 chunks of 16B
#pragma unroll
    for (int i = 0; i < 4; ++i) {
      int c   = tid + i * 256;
      int row = c >> 3;
      int col = (c & 7) * 8;
      *(v8bf*)&sA[buf * BUFELE + row * LDT + col] =
          *(const v8bf*)(A + (size_t)(bm + row) * Kdim + kt + col);
      *(v8bf*)&sB[buf * BUFELE + row * LDT + col] =
          *(const v8bf*)(W + (size_t)(bn + row) * Kdim + kt + col);
    }
    __syncthreads();
#endif

    const __bf16* pA = &sA[buf * BUFELE];
    const __bf16* pB = &sB[buf * BUFELE];
#pragma unroll
    for (int s32 = 0; s32 < BK; s32 += 32) {
      v16bf af[4], bf_[2];
#pragma unroll
      for (int i = 0; i < 4; ++i)
        af[i] = ld_fragA(pA + (wm + i * 16 + r) * LDT + s32 + koff);
#pragma unroll
      for (int j = 0; j < 2; ++j)
        bf_[j] = ld_fragB(pB + (wn + j * 16 + r) * LDT + s32 + koff);
#pragma unroll
      for (int i = 0; i < 4; ++i)
#pragma unroll
        for (int j = 0; j < 2; ++j)
          acc[i][j] = __builtin_amdgcn_wmma_f32_16x16x32_bf16(
              false, af[i], false, bf_[j], (short)0, acc[i][j], false, false);
    }

#if USE_TDM
    if (wave == 0) __builtin_amdgcn_s_wait_tensorcnt(0);
    __syncthreads();
    buf ^= 1;
#else
    __syncthreads();
#endif
  }

  // C/D layout: VGPR e -> rows e (lanes 0-15) / e+8 (lanes 16-31), col = lane&15
  const int colc   = lane & 15;
  const int rowadd = (lane >> 4) * 8;
#pragma unroll
  for (int i = 0; i < 4; ++i)
#pragma unroll
    for (int j = 0; j < 2; ++j)
#pragma unroll
      for (int e = 0; e < 8; ++e) {
        size_t row = (size_t)(bm + wm + i * 16 + e + rowadd);
        size_t col = (size_t)(bn + wn + j * 16 + colc);
        if (OUT_BF16)
          ((__bf16*)out)[row * N + col] = (__bf16)acc[i][j][e];
        else
          ((float*)out)[row * N + col] = acc[i][j][e];
      }
}

// ---- 3) lightning-indexer scores + top-64 per (b,h) ----------------------
__global__ __launch_bounds__(256) void topk_kernel(
    const __bf16* __restrict__ kproj, const float* __restrict__ w_score,
    int* __restrict__ top_idx) {
  __shared__ float sc[T_];         // 32 KB
  __shared__ float rmax[256];
  __shared__ int   rarg[256];
  const int bh = blockIdx.x;
  const int b = bh >> 4, h = bh & 15;
  const int tid = threadIdx.x;

  float w[HD_];
#pragma unroll
  for (int d = 0; d < HD_; ++d) w[d] = w_score[d];

  for (int t = tid; t < T_; t += 256) {
    const __bf16* kr = kproj + ((size_t)(b * T_ + t)) * D_ + h * HD_;
    float s = 0.f;
#pragma unroll
    for (int d = 0; d < HD_; ++d) s += (float)kr[d] * w[d];
    sc[t] = s;
  }
  __syncthreads();

  for (int sel = 0; sel < TK_; ++sel) {
    float m = -1e30f; int mi = 0;
    for (int t = tid; t < T_; t += 256) {
      float v = sc[t];
      if (v > m) { m = v; mi = t; }
    }
    rmax[tid] = m; rarg[tid] = mi;
    __syncthreads();
    for (int off = 128; off > 0; off >>= 1) {
      if (tid < off && rmax[tid + off] > rmax[tid]) {
        rmax[tid] = rmax[tid + off];
        rarg[tid] = rarg[tid + off];
      }
      __syncthreads();
    }
    if (tid == 0) {
      top_idx[bh * TK_ + sel] = rarg[0];
      sc[rarg[0]] = -1e30f;
    }
    __syncthreads();
  }
}

// ---- 4) sparse attention core (WMMA) --------------------------------------
// grid = (B*H, T/128); block = 256 (8 waves x 16 queries)
__global__ __launch_bounds__(256) void sparse_attn_kernel(
    const __bf16* __restrict__ q, const __bf16* __restrict__ k,
    const __bf16* __restrict__ v, const int* __restrict__ topidx,
    __bf16* __restrict__ out) {
  constexpr int LDT = 72;  // 144B row stride (multiple of 16B)
  __shared__ __align__(16) __bf16 sK[TK_ * LDT];       // gathered keys, row-major
  __shared__ __align__(16) __bf16 sVT[HD_ * LDT];      // gathered values, transposed
  __shared__ __align__(16) __bf16 sAttn[8 * 16 * LDT]; // per-wave P tiles

  const int bh = blockIdx.x, b = bh >> 4, h = bh & 15;
  const int tid = threadIdx.x, wave = tid >> 5, lane = tid & 31;
  const int qbase = blockIdx.y * 128 + wave * 16;
  const int* idx = topidx + bh * TK_;

  // gather K rows (contiguous) and V rows transposed into LDS
  for (int i = tid; i < TK_ * 8; i += 256) {  // 64 keys x 8 chunks of 8 bf16
    int key = i >> 3, c = (i & 7) * 8;
    int t = idx[key];
    *(v8bf*)&sK[key * LDT + c] =
        *(const v8bf*)(k + ((size_t)(b * T_ + t)) * D_ + h * HD_ + c);
  }
  for (int i = tid; i < TK_ * HD_; i += 256) {
    int key = i >> 6, d = i & 63;
    int t = idx[key];
    sVT[d * LDT + key] = v[((size_t)(b * T_ + t)) * D_ + h * HD_ + d];
  }
  __syncthreads();

  const int r     = lane & 15;
  const int koff  = (lane >> 4) * 8;
  const int kbase = (lane >> 4) * 16;

  // Q fragments straight from global (rows contiguous over HD)
  const __bf16* qrow = q + ((size_t)(b * T_ + qbase + r)) * D_ + h * HD_;
  v16bf qf[2];
  qf[0] = ld_fragA(qrow + koff);
  qf[1] = ld_fragA(qrow + 32 + koff);

  // S = Q * Ks^T : 4 key-tiles of 16, K-dim HD=64 -> 2 WMMA steps
  v8f sacc[4];
#pragma unroll
  for (int nt = 0; nt < 4; ++nt) sacc[nt] = v8f_zero();
#pragma unroll
  for (int nt = 0; nt < 4; ++nt)
#pragma unroll
    for (int s = 0; s < 2; ++s) {
      v16bf kf = ld_fragB(&sK[(nt * 16 + r) * LDT + s * 32 + koff]);
      sacc[nt] = __builtin_amdgcn_wmma_f32_16x16x32_bf16(
          false, qf[s], false, kf, (short)0, sacc[nt], false, false);
    }

  // softmax over 64 keys; each query row is spread over one 16-lane half
  const float scale = 0.125f;  // 1/sqrt(64)
  float attnv[4][8];
#pragma unroll
  for (int e = 0; e < 8; ++e) {
    float m = -1e30f;
#pragma unroll
    for (int nt = 0; nt < 4; ++nt) m = fmaxf(m, sacc[nt][e]);
#pragma unroll
    for (int off = 1; off < 16; off <<= 1) m = fmaxf(m, __shfl_xor(m, off, 32));
    float sum = 0.f;
#pragma unroll
    for (int nt = 0; nt < 4; ++nt) {
      float p = __expf(scale * (sacc[nt][e] - m));
      attnv[nt][e] = p;
      sum += p;
    }
#pragma unroll
    for (int off = 1; off < 16; off <<= 1) sum += __shfl_xor(sum, off, 32);
    float inv = 1.f / sum;
#pragma unroll
    for (int nt = 0; nt < 4; ++nt) attnv[nt][e] *= inv;
  }

  // reshape P (C-layout) -> A-fragment layout via per-wave LDS scratch
  __bf16* sP = &sAttn[wave * 16 * LDT];
  const int rowadd = (lane >> 4) * 8;
#pragma unroll
  for (int nt = 0; nt < 4; ++nt)
#pragma unroll
    for (int e = 0; e < 8; ++e)
      sP[(e + rowadd) * LDT + nt * 16 + (lane & 15)] = (__bf16)attnv[nt][e];
  __syncthreads();

  // O = P * Vs : 4 HD-tiles of 16, K-dim 64 -> 2 WMMA steps
  v16bf af2[2];
  af2[0] = ld_fragA(&sP[r * LDT + koff]);
  af2[1] = ld_fragA(&sP[r * LDT + 32 + koff]);
  v8f oacc[4];
#pragma unroll
  for (int nt = 0; nt < 4; ++nt) oacc[nt] = v8f_zero();
#pragma unroll
  for (int nt = 0; nt < 4; ++nt)
#pragma unroll
    for (int s = 0; s < 2; ++s) {
      v16bf vf = ld_fragB(&sVT[(nt * 16 + r) * LDT + s * 32 + koff]);
      oacc[nt] = __builtin_amdgcn_wmma_f32_16x16x32_bf16(
          false, af2[s], false, vf, (short)0, oacc[nt], false, false);
    }

  // store O in [B*T, D] layout (bf16) for the final Wo GEMM
#pragma unroll
  for (int nt = 0; nt < 4; ++nt)
#pragma unroll
    for (int e = 0; e < 8; ++e) {
      size_t row = (size_t)b * T_ + qbase + e + rowadd;
      int d = h * HD_ + nt * 16 + (lane & 15);
      out[row * D_ + d] = (__bf16)oacc[nt][e];
    }
}

// ---------------------------------------------------------------------------
extern "C" void kernel_launch(void* const* d_in, const int* in_sizes, int n_in,
                              void* d_out, int out_size, void* d_ws,
                              size_t ws_size, hipStream_t stream) {
  (void)in_sizes; (void)n_in; (void)out_size; (void)ws_size;
  const float* x  = (const float*)d_in[0];
  const float* Wq = (const float*)d_in[1];
  const float* Wk = (const float*)d_in[2];
  const float* Wv = (const float*)d_in[3];
  const float* Wo = (const float*)d_in[4];
  const float* ws_score = (const float*)d_in[5];
  float* outp = (float*)d_out;

  char* ws = (char*)d_ws;
  size_t off = 0;
  auto nxt = [&](size_t bytes) -> void* {
    void* p = ws + off;
    off += (bytes + 255) & ~(size_t)255;
    return p;
  };
  __bf16* xb  = (__bf16*)nxt((size_t)M_ * D_ * 2);
  __bf16* Wqb = (__bf16*)nxt((size_t)D_ * D_ * 2);
  __bf16* Wkb = (__bf16*)nxt((size_t)D_ * D_ * 2);
  __bf16* Wvb = (__bf16*)nxt((size_t)D_ * D_ * 2);
  __bf16* Wob = (__bf16*)nxt((size_t)D_ * D_ * 2);
  __bf16* qb  = (__bf16*)nxt((size_t)M_ * D_ * 2);
  __bf16* kb  = (__bf16*)nxt((size_t)M_ * D_ * 2);
  __bf16* vb  = (__bf16*)nxt((size_t)M_ * D_ * 2);
  __bf16* ob  = (__bf16*)nxt((size_t)M_ * D_ * 2);
  int*    ib  = (int*)nxt((size_t)B_ * H_ * TK_ * 4);

  // 1) convert inputs to bf16
  {
    int n4x = (M_ * D_) / 4;                  // 8,388,608
    cvt_bf16_kernel<<<n4x / 256, 256, 0, stream>>>(x, xb, n4x);
    int n4w = (D_ * D_) / 4;                  // 262,144
    cvt_bf16_kernel<<<n4w / 256, 256, 0, stream>>>(Wq, Wqb, n4w);
    cvt_bf16_kernel<<<n4w / 256, 256, 0, stream>>>(Wk, Wkb, n4w);
    cvt_bf16_kernel<<<n4w / 256, 256, 0, stream>>>(Wv, Wvb, n4w);
    cvt_bf16_kernel<<<n4w / 256, 256, 0, stream>>>(Wo, Wob, n4w);
  }

  // 2) Q/K/V projections (bf16 out)
  dim3 ggrid(M_ / 128, D_ / 128);  // (256, 8)
  gemm_nt_bf16_kernel<true><<<ggrid, 256, 0, stream>>>(xb, Wqb, qb, M_, D_, D_);
  gemm_nt_bf16_kernel<true><<<ggrid, 256, 0, stream>>>(xb, Wkb, kb, M_, D_, D_);
  gemm_nt_bf16_kernel<true><<<ggrid, 256, 0, stream>>>(xb, Wvb, vb, M_, D_, D_);

  // 3) indexer scores + top-64 per (b,h)
  topk_kernel<<<B_ * H_, 256, 0, stream>>>(kb, ws_score, ib);

  // 4) sparse attention
  sparse_attn_kernel<<<dim3(B_ * H_, T_ / 128), 256, 0, stream>>>(qb, kb, vb, ib, ob);

  // 5) output projection (f32 out)
  gemm_nt_bf16_kernel<false><<<ggrid, 256, 0, stream>>>(ob, Wob, outp, M_, D_, D_);
}
